// LocalNormalizedCrossCorrelation_75849122447881
// MI455X (gfx1250) — matrix operand order repaired and checked by reading, not compiled, not run on hardware
//
#include <hip/hip_runtime.h>
#include <stdint.h>

typedef __attribute__((ext_vector_type(16))) _Float16 v16h;
typedef __attribute__((ext_vector_type(8)))  float    v8f;
typedef int v4i_gcc __attribute__((vector_size(16)));                 // matches builtin's pointee
typedef __attribute__((address_space(1))) v4i_gcc* gas_v4i_ptr;       // global (AS1)
typedef __attribute__((address_space(3))) v4i_gcc* lds_v4i_ptr;       // LDS (AS3)

namespace {
constexpr int BB = 2, DD = 160, HH = 192, WWD = 192;
constexpr int TILE = 16, HALO = 4, SPAN = TILE + 2 * HALO;   // 24
constexpr int ROWSTR = 28;                  // 28*l mod 64 distinct (conflict-free) AND rows 16B-aligned
constexpr int NROWS  = SPAN * SPAN;         // 576 (d,h) rows per tile
constexpr int SZ_IN  = NROWS * ROWSTR;      // 16128 floats
constexpr int OFF_F  = 0;
constexpr int OFF_W  = SZ_IN;
constexpr int OFF_T1 = 2 * SZ_IN;           // t1: W-filtered, [R=0..575][16]
constexpr int SZ_T1  = NROWS * TILE;        // 9216
constexpr int OFF_T2 = OFF_T1 + SZ_T1;      // t2: W,H-filtered, [d=0..23][16][16]
constexpr int SZ_T2  = SPAN * TILE * TILE;  // 6144
constexpr int LDS_FLOATS = OFF_T2 + SZ_T2;  // 47616 floats = 190,464 B (<320 KB/WGP)
constexpr size_t LDS_BYTES = (size_t)LDS_FLOATS * sizeof(float);
constexpr int NTILE_D = DD / TILE, NTILE_H = HH / TILE, NTILE_W = WWD / TILE;
constexpr int NBLOCKS = BB * NTILE_D * NTILE_H * NTILE_W;   // 2880
constexpr float EPSV = 1e-8f;
constexpr float INV_WIN = 1.0f / 729.0f;
constexpr float INV_N = 1.0f / (float)((long long)BB * DD * HH * WWD);
}

__device__ __forceinline__ int reflect_idx(int i, int n) {
  if (i < 0) i = -i;                 // jnp 'reflect' (mirror, edge not repeated)
  if (i >= n) i = 2 * n - 2 - i;     // halo 4 << n, single reflection suffices
  return i;
}

__device__ __forceinline__ void wait_async_all() {
#if __has_builtin(__builtin_amdgcn_s_wait_asynccnt)
  __builtin_amdgcn_s_wait_asynccnt(0);
#else
  asm volatile("s_wait_asynccnt 0" ::: "memory");
#endif
}

template <int FLD>
__device__ __forceinline__ float fld_combine(float x, float y) {
  if constexpr (FLD == 0) return x;
  else if constexpr (FLD == 1) return y;
  else if constexpr (FLD == 2) return x * x;
  else if constexpr (FLD == 3) return y * y;
  else return x * y;
}

// One field: W-filter (WMMA) -> H-filter -> D-filter, result in per-thread acc[16].
template <int FLD>
__device__ void filter_field(float* __restrict__ smem, int tid, float* __restrict__ acc) {
  const int lane = tid & 31;
  const int wv   = tid >> 5;       // wave id 0..7 (wave32)
  const int n    = lane & 15;
  const bool hi  = lane >= 16;
  const int kb0  = hi ? 8 : 0;     // K-run for A VGPRs 0..3 (elements 0..7)
  const int kb1  = hi ? 24 : 16;   // K-run for A VGPRs 4..7 (elements 8..15)

  // ---- Stage 1: t1[R][n] = sum_{k=n..n+8} field[R][k] via V_WMMA_F32_16X16X32_F16.
  // Constant banded B (32x16, lane n = column n, VGPR v = K pair): B[k][n] = (n<=k<=n+8).
  // B rows k>=24 are all zero, so garbage A data at K=24..31 (hi lanes) is select-zeroed.
  v16h bmat;
#pragma unroll
  for (int v = 0; v < 8; ++v) {
    int k0 = (hi ? 8 : 0) + (v >= 4 ? 8 : 0) + 2 * v;
    bmat[2 * v]     = (_Float16)((k0     >= n && k0     <= n + 8) ? 1.0f : 0.0f);
    bmat[2 * v + 1] = (_Float16)((k0 + 1 >= n && k0 + 1 <= n + 8) ? 1.0f : 0.0f);
  }
  // 576 rows -> 36 row-groups of 16, round-robin across the 8 waves (uniform per wave).
  for (int t = wv; t < NROWS / 16; t += 8) {
    const int R = t * 16 + n;          // A row M = lane&15
    const float* px = smem + OFF_F + R * ROWSTR;
    const float* py = smem + OFF_W + R * ROWSTR;
    // Unconditional 16B LDS loads (rows are 16B-aligned); unused arrays are dead-code-eliminated.
    float xv[16], yv[16];
    *(float4*)&xv[0]  = *(const float4*)(px + kb0);
    *(float4*)&xv[4]  = *(const float4*)(px + kb0 + 4);
    *(float4*)&xv[8]  = *(const float4*)(px + kb1);
    *(float4*)&xv[12] = *(const float4*)(px + kb1 + 4);
    *(float4*)&yv[0]  = *(const float4*)(py + kb0);
    *(float4*)&yv[4]  = *(const float4*)(py + kb0 + 4);
    *(float4*)&yv[8]  = *(const float4*)(py + kb1);
    *(float4*)&yv[12] = *(const float4*)(py + kb1 + 4);
    v16h amat;                         // element e -> K = (e<8 ? kb0+e : kb1+e-8), ISA A layout
#pragma unroll
    for (int e = 0; e < 16; ++e) {
      float val = fld_combine<FLD>(xv[e], yv[e]);
      val = (hi && e >= 8) ? 0.0f : val;   // branch-free zero of K=24..31 (NaN-safe)
      amat[e] = (_Float16)val;
    }
    v8f c = {};
    c = __builtin_amdgcn_wmma_f32_16x16x32_f16(false, amat, false, bmat, (short)0, c,
                                               false, false);
#pragma unroll
    for (int r = 0; r < 8; ++r) {      // D f32 layout: VGPR r -> M=r (+8 for hi lanes)
      int RR = t * 16 + r + (hi ? 8 : 0);
      smem[OFF_T1 + RR * TILE + n] = c[r];
    }
  }
  __syncthreads();

  // ---- Stage 2: H-axis 9-tap (exact f32): t2[d][h'][w] = sum_k t1[d][h'+k][w]
  for (int i = tid; i < SZ_T2; i += 256) {
    int w = i & 15, hp = (i >> 4) & 15, d = i >> 8;
    float s = 0.0f;
#pragma unroll
    for (int k = 0; k < 9; ++k) s += smem[OFF_T1 + (d * SPAN + hp + k) * TILE + w];
    smem[OFF_T2 + i] = s;
  }
  __syncthreads();

  // ---- Stage 3: D-axis 9-tap into registers; thread owns (hp, w), all 16 dp.
  const int w = tid & 15, hp = tid >> 4;
#pragma unroll
  for (int dp = 0; dp < TILE; ++dp) {
    float s = 0.0f;
#pragma unroll
    for (int k = 0; k < 9; ++k) s += smem[OFF_T2 + ((dp + k) * TILE + hp) * TILE + w];
    acc[dp] = s;
  }
}

__global__ __launch_bounds__(256) void lncc_main(const float* __restrict__ fixedp,
                                                 const float* __restrict__ warpedp,
                                                 float* __restrict__ partial) {
  extern __shared__ float smem[];
  const int tid = threadIdx.x;
  const int b  = blockIdx.z / NTILE_D;
  const int td = blockIdx.z % NTILE_D;
  const int d0 = td * TILE - HALO;
  const int h0 = (int)blockIdx.y * TILE - HALO;
  const int w0 = (int)blockIdx.x * TILE - HALO;
  const size_t volBase = (size_t)b * DD * HH * WWD;

  // Stage 0: load 24^3 haloed tile of both inputs (reflect-padded) into LDS.
  // Fast path (W-interior tiles, contiguous 16B-aligned W-runs): async global->LDS B128
  // (ASYNCcnt-tracked DMA path, no VGPR round trip) when available, else float4 loads.
  if (w0 >= 0 && w0 + SPAN <= WWD) {
    for (int i = tid; i < SPAN * SPAN * (SPAN / 4); i += 256) {   // 24*24*6 float4s
      int wq = i % (SPAN / 4); int r = i / (SPAN / 4);
      int h = r % SPAN; int d = r / SPAN;
      int gd = reflect_idx(d0 + d, DD);
      int gh = reflect_idx(h0 + h, HH);
      size_t g = volBase + ((size_t)gd * HH + gh) * WWD + (w0 + 4 * wq);
      int l = (d * SPAN + h) * ROWSTR + 4 * wq;
#if __has_builtin(__builtin_amdgcn_global_load_async_to_lds_b128)
      // Param0 is a non-const AS1 v4i* (per clang diagnostic); uintptr cast is
      // value-preserving (AS1 uses flat addresses) and drops const.
      __builtin_amdgcn_global_load_async_to_lds_b128(
          (gas_v4i_ptr)(uintptr_t)(fixedp + g),
          (lds_v4i_ptr)(smem + OFF_F + l), 0, 0);
      __builtin_amdgcn_global_load_async_to_lds_b128(
          (gas_v4i_ptr)(uintptr_t)(warpedp + g),
          (lds_v4i_ptr)(smem + OFF_W + l), 0, 0);
#else
      *(float4*)(smem + OFF_F + l) = *(const float4*)(fixedp + g);
      *(float4*)(smem + OFF_W + l) = *(const float4*)(warpedp + g);
#endif
    }
#if __has_builtin(__builtin_amdgcn_global_load_async_to_lds_b128)
    wait_async_all();   // this wave's async LDS writes complete; barrier covers other waves
#endif
  } else {
    for (int i = tid; i < SPAN * SPAN * SPAN; i += 256) {
      int w = i % SPAN; int r = i / SPAN; int h = r % SPAN; int d = r / SPAN;
      int gd = reflect_idx(d0 + d, DD);
      int gh = reflect_idx(h0 + h, HH);
      int gw = reflect_idx(w0 + w, WWD);
      size_t g = volBase + ((size_t)gd * HH + gh) * WWD + gw;
      int l = (d * SPAN + h) * ROWSTR + w;
      smem[OFF_F + l] = fixedp[g];
      smem[OFF_W + l] = warpedp[g];
    }
  }
  __syncthreads();

  float aF[16], aW[16], aFF[16], aWW[16], aFW[16];
  filter_field<0>(smem, tid, aF);
  filter_field<1>(smem, tid, aW);
  filter_field<2>(smem, tid, aFF);
  filter_field<3>(smem, tid, aWW);
  filter_field<4>(smem, tid, aFW);

  // Pointwise LNCC (normalize unnormalized 729-sums here, in f32).
  float lsum = 0.0f;
#pragma unroll
  for (int dp = 0; dp < TILE; ++dp) {
    float muf = aF[dp] * INV_WIN, muw = aW[dp] * INV_WIN;
    float sf2 = fmaxf(aFF[dp] * INV_WIN - muf * muf, EPSV);
    float sw2 = fmaxf(aWW[dp] * INV_WIN - muw * muw, EPSV);
    float sfw = aFW[dp] * INV_WIN - muf * muw;
    float den = fmaxf(sqrtf(fmaxf(sf2 * sw2, EPSV * EPSV)), EPSV);
    float l = sfw / den;
    l = (l == l) ? l : 0.0f;               // nan_to_num (den>0, inputs finite)
    l = fminf(fmaxf(l, -1.0f), 1.0f);      // clip; also maps +/-inf correctly
    lsum += l;
  }

  // Block reduction (reuse front of LDS), deterministic partial per block.
  __syncthreads();
  smem[tid] = lsum;
  __syncthreads();
  for (int s = 128; s > 0; s >>= 1) {
    if (tid < s) smem[tid] += smem[tid + s];
    __syncthreads();
  }
  if (tid == 0) {
    int blk = ((int)blockIdx.z * NTILE_H + (int)blockIdx.y) * NTILE_W + (int)blockIdx.x;
    partial[blk] = smem[0];
  }
}

__global__ __launch_bounds__(256) void lncc_reduce(const float* __restrict__ partial,
                                                   float* __restrict__ out) {
  __shared__ float s[256];
  float a = 0.0f;
  for (int i = threadIdx.x; i < NBLOCKS; i += 256) a += partial[i];
  s[threadIdx.x] = a;
  __syncthreads();
  for (int st = 128; st > 0; st >>= 1) {
    if ((int)threadIdx.x < st) s[threadIdx.x] += s[threadIdx.x + st];
    __syncthreads();
  }
  if (threadIdx.x == 0) out[0] = -s[0] * INV_N;
}

extern "C" void kernel_launch(void* const* d_in, const int* in_sizes, int n_in,
                              void* d_out, int out_size, void* d_ws, size_t ws_size,
                              hipStream_t stream) {
  (void)in_sizes; (void)n_in; (void)out_size; (void)ws_size;
  const float* fixedp  = (const float*)d_in[0];
  const float* warpedp = (const float*)d_in[1];
  float* out     = (float*)d_out;
  float* partial = (float*)d_ws;   // NBLOCKS floats of scratch

  dim3 grid(NTILE_W, NTILE_H, BB * NTILE_D);   // 12 x 12 x 20 = 2880 tiles
  lncc_main<<<grid, 256, LDS_BYTES, stream>>>(fixedp, warpedp, partial);
  lncc_reduce<<<1, 256, 0, stream>>>(partial, out);
}
